// HybridMemoryHead_36661840838806
// MI455X (gfx1250) — compile-verified, matrix operand order checked by reading, not run on hardware
//
#include <hip/hip_runtime.h>
#include <hip/hip_bf16.h>

// ---------------- problem constants (from reference) ----------------
#define B_SAMP   256
#define M_DIM    65536
#define D_DIM    2048
#define C_CLS    8192
#define INV_T    20.0f      // 1 / 0.05
#define EPS_F    1e-6f

// ---------------- GEMM tiling ----------------
#define TILE_N   128        // feature rows per workgroup
#define KC       32         // K per WMMA step (bf16 16x16x32)
#define NSTEP    (D_DIM / KC)
#define AS       40         // A LDS row stride in halfs (32+8 pad -> 80B, conflict-free)
#define BSF      36         // B LDS row stride in dwords (32+4 pad -> 144B, conflict-free)

typedef unsigned short ushort_t;
typedef __attribute__((ext_vector_type(16))) __bf16 v16bf;
typedef __attribute__((ext_vector_type(8)))  float  v8f;

union FragU { v16bf v; uint4 q[2]; };

__device__ __forceinline__ ushort_t f2bf(float f) {
    unsigned u = __float_as_uint(f);
    unsigned r = u + 0x7FFFu + ((u >> 16) & 1u);   // round-to-nearest-even
    return (ushort_t)(r >> 16);
}

// CDNA5 async DMA: global -> LDS, 16B per lane, tracked by ASYNCcnt.
__device__ __forceinline__ void async_b128(unsigned lds_off, const void* gaddr) {
    asm volatile("global_load_async_to_lds_b128 %0, %1, off"
                 :: "v"(lds_off), "v"(gaddr)
                 : "memory");
}
__device__ __forceinline__ void wait_async0() {
    asm volatile("s_wait_asynccnt 0" ::: "memory");
}

// ---------------- kernel 0: zero scratch + output ----------------
__global__ void hmh_zero_ws(float* __restrict__ sim, float* __restrict__ counts,
                            float* __restrict__ out) {
    size_t i = (size_t)blockIdx.x * blockDim.x + threadIdx.x;
    size_t stride = (size_t)gridDim.x * blockDim.x;
    const size_t NS = (size_t)B_SAMP * C_CLS;
    for (size_t k = i; k < NS; k += stride) sim[k] = 0.0f;
    for (size_t k = i; k < C_CLS; k += stride) counts[k] = 0.0f;
    if (i == 0) out[0] = 0.0f;
}

// ---------------- kernel 1: L2-normalize rows of `inputs`, cast to bf16 ----------------
__global__ __launch_bounds__(256) void hmh_normalize_cast(const float* __restrict__ x,
                                                          ushort_t* __restrict__ xn) {
    int b = blockIdx.x, tid = threadIdx.x;
    const float* row = x + (size_t)b * D_DIM;
    float4 v0 = *(const float4*)(row + tid * 8);
    float4 v1 = *(const float4*)(row + tid * 8 + 4);
    float s = v0.x * v0.x + v0.y * v0.y + v0.z * v0.z + v0.w * v0.w
            + v1.x * v1.x + v1.y * v1.y + v1.z * v1.z + v1.w * v1.w;
    __shared__ float red[256];
    red[tid] = s;
    __syncthreads();
    for (int st = 128; st > 0; st >>= 1) {
        if (tid < st) red[tid] += red[tid + st];
        __syncthreads();
    }
    float rn = rsqrtf(red[0]);
    union { ushort_t h[8]; uint4 q; } o;
    o.h[0] = f2bf(v0.x * rn); o.h[1] = f2bf(v0.y * rn);
    o.h[2] = f2bf(v0.z * rn); o.h[3] = f2bf(v0.w * rn);
    o.h[4] = f2bf(v1.x * rn); o.h[5] = f2bf(v1.y * rn);
    o.h[6] = f2bf(v1.z * rn); o.h[7] = f2bf(v1.w * rn);
    *(uint4*)(xn + (size_t)b * D_DIM + tid * 8) = o.q;
}

// ---------------- kernel 2: label counts ----------------
__global__ void hmh_count_labels(const int* __restrict__ labels, float* __restrict__ counts) {
    int m = blockIdx.x * blockDim.x + threadIdx.x;
    if (m < M_DIM) atomicAdd(&counts[labels[m]], 1.0f);
}

// ---------------- kernel 3: WMMA GEMM + label scatter, async ping-pong LDS ----------------
// 256 threads = 8 waves. Tile: 256(B) x 128(m), K-loop over D=2048 in 32-steps.
// Next K-tile is DMA'd via global_load_async_to_lds_b128 while WMMAs run on the
// current tile; one s_wait_asynccnt + one barrier per step.
__global__ __launch_bounds__(256) void hmh_gemm_scatter(const ushort_t* __restrict__ xn,
                                                        const float* __restrict__ feat,
                                                        const int* __restrict__ labels,
                                                        float* __restrict__ sim) {
    __shared__ __align__(16) ushort_t aT[2][B_SAMP * AS];   // 2 x 20 KB, bf16
    __shared__ __align__(16) float    bT[2][TILE_N * BSF];  // 2 x 18 KB, fp32

    const int tid  = threadIdx.x;
    const int wave = tid >> 5;
    const int lane = tid & 31;
    const int lr   = lane & 15;              // row within 16x16 tile
    const int kb   = (lane < 16) ? 0 : 8;    // K sub-base per documented 16-bit A layout
    const int m0   = blockIdx.x * TILE_N;

    // per-thread staging coordinates (fixed across K-steps)
    //  A: 256 rows x 2 chunks of 16 halfs? -> 256 rows x 4 chunks of 8 halfs (16B)
    //  B: 128 rows x 8 chunks of 4 floats (16B)
    const unsigned aBase0 = (unsigned)(uintptr_t)&aT[0][0];
    const unsigned aBase1 = (unsigned)(uintptr_t)&aT[1][0];
    const unsigned bBase0 = (unsigned)(uintptr_t)&bT[0][0];
    const unsigned bBase1 = (unsigned)(uintptr_t)&bT[1][0];

    v8f acc[16];
#pragma unroll
    for (int t = 0; t < 16; ++t) acc[t] = (v8f){0, 0, 0, 0, 0, 0, 0, 0};

    // ---- stage K-step `kk` into buffer `buf` (async DMA, no VGPR round-trip) ----
    auto stage = [&](int buf, int kk) {
        unsigned aBase = buf ? aBase1 : aBase0;
        unsigned bBase = buf ? bBase1 : bBase0;
#pragma unroll
        for (int p = 0; p < 4; ++p) {                 // A: 1024 x 16B chunks / 256 thr
            int lin = p * 256 + tid;
            int row = lin >> 2, ch = (lin & 3) * 8;   // ch in halfs
            async_b128(aBase + row * (AS * 2) + ch * 2,
                       xn + (size_t)row * D_DIM + kk + ch);
        }
#pragma unroll
        for (int p = 0; p < 4; ++p) {                 // B: 1024 x 16B chunks / 256 thr
            int lin = p * 256 + tid;
            int row = lin >> 3, ch = (lin & 7) * 4;   // ch in floats
            async_b128(bBase + row * (BSF * 4) + ch * 4,
                       feat + (size_t)(m0 + row) * D_DIM + kk + ch);
        }
    };

    stage(0, 0);
    wait_async0();
    __syncthreads();

    for (int step = 0; step < NSTEP; ++step) {
        const int cur = step & 1;
        if (step + 1 < NSTEP) stage(cur ^ 1, (step + 1) * KC);

        // B fragment: read fp32 from LDS, convert to bf16 (used by 16 WMMAs)
        const float* bp = &bT[cur][(wave * 16 + lr) * BSF];
        float4 q0 = *(const float4*)(bp + kb);
        float4 q1 = *(const float4*)(bp + kb + 4);
        float4 q2 = *(const float4*)(bp + kb + 16);
        float4 q3 = *(const float4*)(bp + kb + 20);
        union { ushort_t h[16]; v16bf v; } bw;
        bw.h[0]  = f2bf(q0.x); bw.h[1]  = f2bf(q0.y); bw.h[2]  = f2bf(q0.z); bw.h[3]  = f2bf(q0.w);
        bw.h[4]  = f2bf(q1.x); bw.h[5]  = f2bf(q1.y); bw.h[6]  = f2bf(q1.z); bw.h[7]  = f2bf(q1.w);
        bw.h[8]  = f2bf(q2.x); bw.h[9]  = f2bf(q2.y); bw.h[10] = f2bf(q2.z); bw.h[11] = f2bf(q2.w);
        bw.h[12] = f2bf(q3.x); bw.h[13] = f2bf(q3.y); bw.h[14] = f2bf(q3.z); bw.h[15] = f2bf(q3.w);

        // A fragments: software-pipelined (load t+1 while WMMA t)
        const ushort_t* abuf = &aT[cur][0];
        FragU a0, a1;
        {
            const ushort_t* ap = abuf + (0 * 16 + lr) * AS + kb;
            a0.q[0] = *(const uint4*)ap;
            a0.q[1] = *(const uint4*)(ap + 16);
        }
#pragma unroll
        for (int t = 0; t < 16; ++t) {
            FragU& ac = (t & 1) ? a1 : a0;
            FragU& an = (t & 1) ? a0 : a1;
            if (t < 15) {
                const ushort_t* ap = abuf + ((t + 1) * 16 + lr) * AS + kb;
                an.q[0] = *(const uint4*)ap;
                an.q[1] = *(const uint4*)(ap + 16);
            }
            acc[t] = __builtin_amdgcn_wmma_f32_16x16x32_bf16(
                false, ac.v, false, bw.v, (short)0, acc[t], false, false);
        }

        wait_async0();      // next buffer fully landed (this wave's DMAs)
        __syncthreads();    // all waves done reading cur + all DMAs visible
    }

    // epilogue: scatter logits/T into per-class sums
    const int mcol = m0 + wave * 16 + lr;
    const int cls  = labels[mcol];
    const int rofs = (lane >> 4) << 3;    // lanes 16..31 hold rows M+8
#pragma unroll
    for (int t = 0; t < 16; ++t) {
#pragma unroll
        for (int r = 0; r < 8; ++r) {
            int brow = t * 16 + rofs + r;
            atomicAdd(&sim[(size_t)brow * C_CLS + cls], acc[t][r] * INV_T);
        }
    }
}

// ---------------- kernel 4: masked softmax over classes + NLL ----------------
__global__ __launch_bounds__(256) void hmh_loss(const float* __restrict__ sim,
                                                const float* __restrict__ counts,
                                                const int* __restrict__ labels,
                                                const int* __restrict__ idx,
                                                float* __restrict__ out) {
    int b = blockIdx.x, tid = threadIdx.x;
    int tgt = labels[idx[b]];
    const float* srow = sim + (size_t)b * C_CLS;
    float lsum = 0.0f, lpick = 0.0f;
    for (int c = tid; c < C_CLS; c += 256) {
        float cnt = counts[c];
        if (cnt > 0.0f) {
            float e = expf(srow[c] / cnt);
            lsum += e;
            if (c == tgt) lpick = e;
        }
    }
    __shared__ float rs[256], rp[256];
    rs[tid] = lsum; rp[tid] = lpick;
    __syncthreads();
    for (int st = 128; st > 0; st >>= 1) {
        if (tid < st) { rs[tid] += rs[tid + st]; rp[tid] += rp[tid + st]; }
        __syncthreads();
    }
    if (tid == 0) {
        float masked = rp[0] / (rs[0] + EPS_F);
        atomicAdd(out, -logf(masked + EPS_F) * (1.0f / (float)B_SAMP));
    }
}

// ---------------- launcher ----------------
extern "C" void kernel_launch(void* const* d_in, const int* in_sizes, int n_in,
                              void* d_out, int out_size, void* d_ws, size_t ws_size,
                              hipStream_t stream) {
    const float* inputs   = (const float*)d_in[0];   // [256, 2048]
    const float* features = (const float*)d_in[1];   // [65536, 2048] (pre-normalized)
    const int*   labels   = (const int*)d_in[2];     // [65536]
    const int*   idx      = (const int*)d_in[3];     // [256]
    float*       out      = (float*)d_out;           // scalar loss

    char* ws = (char*)d_ws;
    const size_t XN_BYTES  = (size_t)B_SAMP * D_DIM * sizeof(ushort_t);  // 1 MB
    const size_t SIM_BYTES = (size_t)B_SAMP * C_CLS * sizeof(float);     // 8 MB
    ushort_t* xn     = (ushort_t*)ws;
    float*    sim    = (float*)(ws + XN_BYTES);
    float*    counts = (float*)(ws + XN_BYTES + SIM_BYTES);

    hmh_zero_ws<<<2048, 256, 0, stream>>>(sim, counts, out);
    hmh_normalize_cast<<<B_SAMP, 256, 0, stream>>>(inputs, xn);
    hmh_count_labels<<<M_DIM / 256, 256, 0, stream>>>(labels, counts);
    hmh_gemm_scatter<<<M_DIM / TILE_N, 256, 0, stream>>>(xn, features, labels, sim);
    hmh_loss<<<B_SAMP, 256, 0, stream>>>(sim, counts, labels, idx, out);
}
